// HeteroGNN_4879082848459
// MI455X (gfx1250) — compile-verified, matrix operand order checked by reading, not run on hardware
//
#include <hip/hip_runtime.h>

// ---------------------------------------------------------------------------
// Types for CDNA5 WMMA (wave32, 16x16x32 bf16 -> f32)
// ---------------------------------------------------------------------------
typedef __bf16 v16bf __attribute__((ext_vector_type(16)));
typedef __bf16 v8bf  __attribute__((ext_vector_type(8)));
typedef float  v8f   __attribute__((ext_vector_type(8)));

union FragU { v16bf v; v8bf h[2]; };
union Pack8 { v8bf v; __bf16 e[8]; };

#define Hdim 512
#define SLOPE 0.2f
#define LAMW 0.5f

__device__ __forceinline__ float lrelu(float x) { return x > 0.f ? x : SLOPE * x; }

// ---------------------------------------------------------------------------
// Elementwise / utility kernels
// ---------------------------------------------------------------------------
__global__ void fill_i32(int* __restrict__ p, long long n, int v) {
    long long i = (long long)blockIdx.x * blockDim.x + threadIdx.x;
    if (i < n) p[i] = v;
}

__global__ void copy_i32(int* __restrict__ dst, const int* __restrict__ src, long long n) {
    long long i = (long long)blockIdx.x * blockDim.x + threadIdx.x;
    if (i < n) dst[i] = src[i];
}

// dst[i] = lrelu(alpha * src[i])
__global__ void scale_lrelu_k(float* __restrict__ dst, const float* __restrict__ src,
                              float alpha, long long n) {
    long long i = (long long)blockIdx.x * blockDim.x + threadIdx.x;
    if (i < n) dst[i] = lrelu(alpha * src[i]);
}

// dst[i] = a[i] + lam * b[i]
__global__ void axpy_k(float* __restrict__ dst, const float* __restrict__ a,
                       const float* __restrict__ b, float lam, long long n) {
    long long i = (long long)blockIdx.x * blockDim.x + threadIdx.x;
    if (i < n) dst[i] = a[i] + lam * b[i];
}

// Convert W[K x N] fp32 -> Bt[Npad x Kpad] bf16, transposed, zero padded.
__global__ void convert_wT(const float* __restrict__ W, __bf16* __restrict__ Bt,
                           int K, int N, int Kpad, int Npad) {
    long long i = (long long)blockIdx.x * blockDim.x + threadIdx.x;
    long long tot = (long long)Npad * Kpad;
    if (i >= tot) return;
    int n = (int)(i / Kpad);
    int k = (int)(i % Kpad);
    float v = (n < N && k < K) ? W[(long long)k * N + n] : 0.0f;
    Bt[i] = (__bf16)v;
}

// Convert X[M x K] fp32 -> Xb[M x Kpad] bf16, zero padded K tail.
__global__ void convert_a_bf16(const float* __restrict__ X, __bf16* __restrict__ Xb,
                               int K, int Kpad, long long tot) {
    long long i = (long long)blockIdx.x * blockDim.x + threadIdx.x;
    if (i >= tot) return;
    long long m = i / Kpad;
    int k = (int)(i % Kpad);
    Xb[i] = (__bf16)((k < K) ? X[m * K + k] : 0.0f);
}

// ---------------------------------------------------------------------------
// WMMA GEMM: C[M,N] (fp32, ld=N) = A[M,K] @ Bt^T  (+bias)(+C)(lrelu)
//   ABF16=0: A fp32 [M x K].  ABF16=1: A bf16 [M x Kpad] (pre-padded).
//   Bt bf16 transposed [Npad x Kpad], Npad mult 256, Kpad mult 32.
// Block: 256 threads = 8 waves (2x4), block tile 128x256, wave tile 64x64.
// ---------------------------------------------------------------------------
#define LDA_S 40   // 32 + 8 bf16 padding; row stride 80B (16B aligned)

template <int ABF16>
__global__ __launch_bounds__(256) void gemm_wmma_k(
    const void* __restrict__ Aptr, const __bf16* __restrict__ Bt,
    const float* __restrict__ bias, float* __restrict__ C,
    int M, int N, int K, int Kpad, int accum, int act) {

    __shared__ __bf16 As[128 * LDA_S];

    const int tid  = threadIdx.x;
    const int wave = tid >> 5;
    const int lane = tid & 31;
    const int laneLo  = lane & 15;
    const int laneHi8 = (lane >> 4) << 3;   // 0 or 8

    const int m0 = blockIdx.y * 128;
    const int n0 = blockIdx.x * 256;
    const int wmBase = (wave >> 2) * 64;    // 0 or 64
    const int wnBase = (wave & 3) * 64;     // 0..192

    v8f acc[4][4];
#pragma unroll
    for (int i = 0; i < 4; ++i)
#pragma unroll
        for (int j = 0; j < 4; ++j) { v8f zz = {}; acc[i][j] = zz; }

    // staging assignment: thread -> (row sr, 16-wide k-half sk)
    const int sr = tid >> 1;             // 0..127
    const int sk = (tid & 1) << 4;       // 0 or 16
    const int gm = m0 + sr;
    const bool rowOK = (gm < M);
    const bool mFull = (m0 + 128 <= M);
    __bf16* sdst = &As[sr * LDA_S + sk];

    for (int kb = 0; kb < Kpad; kb += 32) {
        __syncthreads();
        if (ABF16) {
            // A pre-padded bf16: two aligned 16B loads, guard rows only
            const __bf16* ap = (const __bf16*)Aptr + (long long)gm * Kpad + kb + sk;
            v8bf lo = {}, hi = {};
            if (rowOK) { lo = *(const v8bf*)ap; hi = *(const v8bf*)(ap + 8); }
            *(v8bf*)sdst = lo;
            *(v8bf*)(sdst + 8) = hi;
        } else {
            const float* ap = (const float*)Aptr + (long long)gm * K + kb + sk;
            if (mFull && ((K & 3) == 0) && (kb + 32 <= K)) {
                // uniform fast path: vectorized fp32 loads + packed converts
                float t[16];
                *(float4*)(t + 0)  = *(const float4*)(ap + 0);
                *(float4*)(t + 4)  = *(const float4*)(ap + 4);
                *(float4*)(t + 8)  = *(const float4*)(ap + 8);
                *(float4*)(t + 12) = *(const float4*)(ap + 12);
                Pack8 lo, hi;
#pragma unroll
                for (int i = 0; i < 8; ++i) { lo.e[i] = (__bf16)t[i]; hi.e[i] = (__bf16)t[8 + i]; }
                *(v8bf*)sdst = lo.v;
                *(v8bf*)(sdst + 8) = hi.v;
            } else {
                Pack8 lo, hi;
#pragma unroll
                for (int i = 0; i < 8; ++i) {
                    int k0 = kb + sk + i, k1 = k0 + 8;
                    lo.e[i] = (__bf16)((rowOK && k0 < K) ? ap[i] : 0.0f);
                    hi.e[i] = (__bf16)((rowOK && k1 < K) ? ap[8 + i] : 0.0f);
                }
                *(v8bf*)sdst = lo.v;
                *(v8bf*)(sdst + 8) = hi.v;
            }
        }
        __syncthreads();

        v16bf afr[4], bfr[4];
#pragma unroll
        for (int ti = 0; ti < 4; ++ti) {
            const __bf16* rp = &As[(wmBase + ti * 16 + laneLo) * LDA_S];
            FragU u;
            u.h[0] = *(const v8bf*)(rp + laneHi8);
            u.h[1] = *(const v8bf*)(rp + 16 + laneHi8);
            afr[ti] = u.v;
        }
#pragma unroll
        for (int tn = 0; tn < 4; ++tn) {
            const __bf16* bp = Bt + (long long)(n0 + wnBase + tn * 16 + laneLo) * Kpad + kb;
            FragU u;
            u.h[0] = *(const v8bf*)(bp + laneHi8);
            u.h[1] = *(const v8bf*)(bp + 16 + laneHi8);
            bfr[tn] = u.v;
            __builtin_prefetch(bp + 32, 0, 0);
        }
#pragma unroll
        for (int ti = 0; ti < 4; ++ti)
#pragma unroll
            for (int tn = 0; tn < 4; ++tn)
                acc[ti][tn] = __builtin_amdgcn_wmma_f32_16x16x32_bf16(
                    false, afr[ti], false, bfr[tn], (short)0, acc[ti][tn], false, false);
    }

    // epilogue
#pragma unroll
    for (int ti = 0; ti < 4; ++ti) {
#pragma unroll
        for (int tn = 0; tn < 4; ++tn) {
            const int n = n0 + wnBase + tn * 16 + laneLo;
#pragma unroll
            for (int r = 0; r < 8; ++r) {
                const int m = m0 + wmBase + ti * 16 + laneHi8 + r;
                if (m < M && n < N) {
                    float v = acc[ti][tn][r];
                    if (bias)  v += bias[n];
                    if (accum) v += C[(long long)m * N + n];
                    if (act)   v = lrelu(v);
                    C[(long long)m * N + n] = v;
                }
            }
        }
    }
}

// ---------------------------------------------------------------------------
// CSR build (deterministic after per-node sort) + gather kernels
// ---------------------------------------------------------------------------
__global__ void deg_count_k(const int* __restrict__ edst, int* __restrict__ deg, int E) {
    int e = blockIdx.x * blockDim.x + threadIdx.x;
    if (e < E) atomicAdd(&deg[edst[e]], 1);
}

// single block, 1024 threads: exclusive scan deg[0..n) -> off[0..n]
__global__ __launch_bounds__(1024) void scan_offsets_k(const int* __restrict__ deg,
                                                       int* __restrict__ off, int n) {
    __shared__ int part[1024];
    const int t = threadIdx.x;
    const int chunk = (n + 1023) / 1024;
    const int begin = t * chunk;
    const int end = (begin + chunk < n) ? begin + chunk : n;
    int s = 0;
    for (int i = begin; i < end; ++i) s += deg[i];
    part[t] = s;
    __syncthreads();
    for (int d = 1; d < 1024; d <<= 1) {
        int v = (t >= d) ? part[t - d] : 0;
        __syncthreads();
        part[t] += v;
        __syncthreads();
    }
    int run = (t == 0) ? 0 : part[t - 1];
    for (int i = begin; i < end; ++i) { off[i] = run; run += deg[i]; }
    if (t == 1023) off[n] = part[1023];
}

__global__ void csr_fill_k(const int* __restrict__ esrc, const int* __restrict__ edst,
                           int* __restrict__ cursor, int* __restrict__ srcArr, int E) {
    int e = blockIdx.x * blockDim.x + threadIdx.x;
    if (e < E) {
        int pos = atomicAdd(&cursor[edst[e]], 1);
        srcArr[pos] = esrc[e];
    }
}

// per-node insertion sort -> deterministic reduction order
__global__ void csr_sort_k(const int* __restrict__ off, int* __restrict__ srcArr, int n_dst) {
    int d = blockIdx.x * blockDim.x + threadIdx.x;
    if (d >= n_dst) return;
    int j0 = off[d], j1 = off[d + 1];
    for (int a = j0 + 1; a < j1; ++a) {
        int v = srcArr[a];
        int b = a - 1;
        while (b >= j0 && srcArr[b] > v) { srcArr[b + 1] = srcArr[b]; --b; }
        srcArr[b + 1] = v;
    }
}

// segment mean via gather: one block (128 thr = float4 cols) per dst node
__global__ __launch_bounds__(128) void gather_mean_k(
    const float* __restrict__ x, const int* __restrict__ off,
    const int* __restrict__ srcArr, float* __restrict__ agg) {
    __shared__ int sIdx[128];
    const int d = blockIdx.x;
    const int c = threadIdx.x * 4;
    const int j0 = off[d], j1 = off[d + 1];
    float4 acc = {0.f, 0.f, 0.f, 0.f};
    for (int jc = j0; jc < j1; jc += 128) {
        int cntc = j1 - jc; if (cntc > 128) cntc = 128;
        if ((int)threadIdx.x < cntc) sIdx[threadIdx.x] = srcArr[jc + threadIdx.x];
        __syncthreads();
        for (int j = 0; j < cntc; ++j) {
            const float4 v = *(const float4*)(x + (long long)sIdx[j] * Hdim + c);
            acc.x += v.x; acc.y += v.y; acc.z += v.z; acc.w += v.w;
        }
        __syncthreads();
    }
    const float inv = 1.0f / fmaxf((float)(j1 - j0), 1.0f);
    float4 r = {acc.x * inv, acc.y * inv, acc.z * inv, acc.w * inv};
    *(float4*)(agg + (long long)d * Hdim + c) = r;
}

// ---------------------------------------------------------------------------
// CSRA pooling (CSR gather form)
// ---------------------------------------------------------------------------
__global__ void score_rows_k(const float* __restrict__ h, const float* __restrict__ w,
                             float* __restrict__ score, int n) {
    const int row  = blockIdx.x * 8 + (threadIdx.x >> 5);
    const int lane = threadIdx.x & 31;
    if (row >= n) return;
    const float* p = h + (long long)row * Hdim;
    float s = 0.f;
    for (int c = lane * 4; c < Hdim; c += 128) {
        float4 v = *(const float4*)(p + c);
        float4 wv = *(const float4*)(w + c);
        s += v.x * wv.x + v.y * wv.y + v.z * wv.z + v.w * wv.w;
    }
#pragma unroll
    for (int o = 16; o > 0; o >>= 1) s += __shfl_down(s, o);
    if (lane == 0) score[row] = s;
}

// per dst node: softmax attention weights over its refer edges
__global__ void csra_att_k(const float* __restrict__ score, const int* __restrict__ off,
                           const int* __restrict__ srcArr, float* __restrict__ att, int n_dst) {
    int d = blockIdx.x * blockDim.x + threadIdx.x;
    if (d >= n_dst) return;
    int j0 = off[d], j1 = off[d + 1];
    float m = -3.4e38f;
    for (int j = j0; j < j1; ++j) m = fmaxf(m, score[srcArr[j]]);
    float z = 0.f;
    for (int j = j0; j < j1; ++j) z += __expf(score[srcArr[j]] - m);
    z = fmaxf(z, 1e-12f);
    for (int j = j0; j < j1; ++j) att[j] = __expf(score[srcArr[j]] - m) / z;
}

// pooled[d] = sum_j att[j] * h[src[j]]   (gather, no atomics)
__global__ __launch_bounds__(128) void pooled_gather_k(
    const float* __restrict__ h, const int* __restrict__ off,
    const int* __restrict__ srcArr, const float* __restrict__ att,
    float* __restrict__ pooled) {
    __shared__ int sIdx[128];
    __shared__ float sW[128];
    const int d = blockIdx.x;
    const int c = threadIdx.x * 4;
    const int j0 = off[d], j1 = off[d + 1];
    float4 acc = {0.f, 0.f, 0.f, 0.f};
    for (int jc = j0; jc < j1; jc += 128) {
        int cntc = j1 - jc; if (cntc > 128) cntc = 128;
        if ((int)threadIdx.x < cntc) {
            sIdx[threadIdx.x] = srcArr[jc + threadIdx.x];
            sW[threadIdx.x] = att[jc + threadIdx.x];
        }
        __syncthreads();
        for (int j = 0; j < cntc; ++j) {
            const float w = sW[j];
            const float4 v = *(const float4*)(h + (long long)sIdx[j] * Hdim + c);
            acc.x += w * v.x; acc.y += w * v.y; acc.z += w * v.z; acc.w += w * v.w;
        }
        __syncthreads();
    }
    *(float4*)(pooled + (long long)d * Hdim + c) = acc;
}

// ---------------------------------------------------------------------------
// Host launcher
// ---------------------------------------------------------------------------
static inline long long cdiv(long long a, long long b) { return (a + b - 1) / b; }

extern "C" void kernel_launch(void* const* d_in, const int* in_sizes, int n_in,
                              void* d_out, int out_size, void* d_ws, size_t ws_size,
                              hipStream_t stream) {
    if (n_in < 21) return;
    const int NW = in_sizes[0] / 1949;
    const int NE = in_sizes[1] / 100;
    const int E  = in_sizes[2] / 2;
    const int L  = 4;
    const int Hn = Hdim;
    const int NMAX = (NW > NE) ? NW : NE;

    const float* x_window = (const float*)d_in[0];
    const float* x_example = (const float*)d_in[1];
    const int* e_near  = (const int*)d_in[2];
    const int* e_close = (const int*)d_in[3];
    const int* e_refer = (const int*)d_in[4];
    const float* W_win  = (const float*)d_in[5];
    const float* W_exp  = (const float*)d_in[6];
    const float* W_post = (const float*)d_in[7];
    const float* Wl_near = (const float*)d_in[9];
    const float* Wr_near = (const float*)d_in[10];
    const float* b_near  = (const float*)d_in[11];
    const float* Wl_close = (const float*)d_in[12];
    const float* Wr_close = (const float*)d_in[13];
    const float* b_close  = (const float*)d_in[14];
    const float* Wl_refer = (const float*)d_in[15];
    const float* Wr_refer = (const float*)d_in[16];
    const float* b_refer  = (const float*)d_in[17];
    const float* w_pool = (const float*)d_in[18];
    const float* W_lin  = (const float*)d_in[19];
    const float* b_lin  = (const float*)d_in[20];

    // ---- workspace carve ----
    char* base = (char*)d_ws; size_t off = 0;
    auto alloc = [&](size_t bytes) -> void* {
        void* r = base + off; off = (off + bytes + 255) & ~(size_t)255; return r;
    };
    float* h_win = (float*)alloc((size_t)NW * Hn * 4);
    float* h_exp = (float*)alloc((size_t)NE * Hn * 4);
    float* o_win = (float*)alloc((size_t)NW * Hn * 4);
    float* o_cls = (float*)alloc((size_t)NE * Hn * 4);
    float* T1    = (float*)alloc((size_t)NMAX * Hn * 4);
    float* score = (float*)alloc((size_t)NE * 4);
    float* attE  = (float*)alloc((size_t)E * 4);
    int* deg     = (int*)alloc((size_t)NMAX * 4);
    int* cursor  = (int*)alloc((size_t)NMAX * 4);
    int* off_nr  = (int*)alloc((size_t)(NW + 1) * 4);
    int* src_nr  = (int*)alloc((size_t)E * 4);
    int* off_rf  = (int*)alloc((size_t)(NW + 1) * 4);
    int* src_rf  = (int*)alloc((size_t)E * 4);
    int* off_cl  = (int*)alloc((size_t)(NE + 1) * 4);
    int* src_cl  = (int*)alloc((size_t)E * 4);
    __bf16* wt_win  = (__bf16*)alloc((size_t)512 * 1952 * 2);
    __bf16* wt_exp  = (__bf16*)alloc((size_t)512 * 128 * 2);
    __bf16* wt_post = (__bf16*)alloc((size_t)512 * 512 * 2);
    __bf16* wt_nl   = (__bf16*)alloc((size_t)L * 512 * 512 * 2);
    __bf16* wt_nr   = (__bf16*)alloc((size_t)L * 512 * 512 * 2);
    __bf16* wt_cl   = (__bf16*)alloc((size_t)L * 512 * 512 * 2);
    __bf16* wt_cr   = (__bf16*)alloc((size_t)L * 512 * 512 * 2);
    __bf16* wt_rl   = (__bf16*)alloc((size_t)L * 512 * 512 * 2);
    __bf16* wt_rr   = (__bf16*)alloc((size_t)L * 512 * 512 * 2);
    __bf16* wt_lin  = (__bf16*)alloc((size_t)256 * 512 * 2);
    // optional bf16 padded copies of the inputs (only if workspace allows)
    __bf16* xe_b = nullptr;
    if (off + (size_t)NE * 128 * 2 + 256 <= ws_size) xe_b = (__bf16*)alloc((size_t)NE * 128 * 2);
    __bf16* xw_b = nullptr;
    if (off + (size_t)NW * 1952 * 2 + 256 <= ws_size) xw_b = (__bf16*)alloc((size_t)NW * 1952 * 2);
    float* pooled = o_win;  // reuse after layers

    auto conv = [&](const float* W, __bf16* Bt, int K, int N, int Kpad, int Npad) {
        long long tot = (long long)Npad * Kpad;
        convert_wT<<<dim3((unsigned)cdiv(tot, 256)), dim3(256), 0, stream>>>(W, Bt, K, N, Kpad, Npad);
    };
    auto gemm = [&](const void* A, int abf16, const __bf16* Bt, const float* bias, float* C,
                    int M, int N, int K, int Kpad, int accum, int act) {
        dim3 g((unsigned)cdiv(N, 256), (unsigned)cdiv(M, 128));
        if (abf16)
            gemm_wmma_k<1><<<g, dim3(256), 0, stream>>>(A, Bt, bias, C, M, N, K, Kpad, accum, act);
        else
            gemm_wmma_k<0><<<g, dim3(256), 0, stream>>>(A, Bt, bias, C, M, N, K, Kpad, accum, act);
    };
    auto build_csr = [&](const int* esrc, const int* edst, int n_dst, int* offp, int* srcArr) {
        fill_i32<<<dim3((unsigned)cdiv(n_dst, 256)), dim3(256), 0, stream>>>(deg, n_dst, 0);
        deg_count_k<<<dim3((unsigned)cdiv(E, 256)), dim3(256), 0, stream>>>(edst, deg, E);
        scan_offsets_k<<<dim3(1), dim3(1024), 0, stream>>>(deg, offp, n_dst);
        copy_i32<<<dim3((unsigned)cdiv(n_dst, 256)), dim3(256), 0, stream>>>(cursor, offp, n_dst);
        csr_fill_k<<<dim3((unsigned)cdiv(E, 256)), dim3(256), 0, stream>>>(esrc, edst, cursor, srcArr, E);
        csr_sort_k<<<dim3((unsigned)cdiv(n_dst, 256)), dim3(256), 0, stream>>>(offp, srcArr, n_dst);
    };
    auto seg_mean = [&](const float* x, const int* offp, const int* srcArr, int n_dst, float* agg) {
        gather_mean_k<<<dim3((unsigned)n_dst), dim3(128), 0, stream>>>(x, offp, srcArr, agg);
    };

    // ---- CSR build (edges constant across layers) ----
    build_csr(e_near, e_near + E, NW, off_nr, src_nr);
    build_csr(e_refer, e_refer + E, NW, off_rf, src_rf);
    build_csr(e_close, e_close + E, NE, off_cl, src_cl);

    // ---- weight conversion (bf16, transposed, padded) ----
    conv(W_win, wt_win, 1949, 512, 1952, 512);
    conv(W_exp, wt_exp, 100, 512, 128, 512);
    conv(W_post, wt_post, 512, 512, 512, 512);
    for (int l = 0; l < L; ++l) {
        const size_t wo = (size_t)l * 512 * 512;
        conv(Wl_near + wo,  wt_nl + wo, 512, 512, 512, 512);
        conv(Wr_near + wo,  wt_nr + wo, 512, 512, 512, 512);
        conv(Wl_close + wo, wt_cl + wo, 512, 512, 512, 512);
        conv(Wr_close + wo, wt_cr + wo, 512, 512, 512, 512);
        conv(Wl_refer + wo, wt_rl + wo, 512, 512, 512, 512);
        conv(Wr_refer + wo, wt_rr + wo, 512, 512, 512, 512);
    }
    conv(W_lin, wt_lin, 512, 100, 512, 256);

    // ---- input projections: h = lrelu(lrelu(x @ W) @ W_post) ----
    if (xe_b) {
        convert_a_bf16<<<dim3((unsigned)cdiv((long long)NE * 128, 256)), dim3(256), 0, stream>>>(
            x_example, xe_b, 100, 128, (long long)NE * 128);
        gemm(xe_b, 1, wt_exp, nullptr, T1, NE, 512, 100, 128, 0, 1);
    } else {
        gemm(x_example, 0, wt_exp, nullptr, T1, NE, 512, 100, 128, 0, 1);
    }
    gemm(T1, 0, wt_post, nullptr, h_exp, NE, 512, 512, 512, 0, 1);
    if (xw_b) {
        convert_a_bf16<<<dim3((unsigned)cdiv((long long)NW * 1952, 256)), dim3(256), 0, stream>>>(
            x_window, xw_b, 1949, 1952, (long long)NW * 1952);
        gemm(xw_b, 1, wt_win, nullptr, T1, NW, 512, 1949, 1952, 0, 1);
    } else {
        gemm(x_window, 0, wt_win, nullptr, T1, NW, 512, 1949, 1952, 0, 1);
    }
    gemm(T1, 0, wt_post, nullptr, h_win, NW, 512, 512, 512, 0, 1);

    // ---- 4 SAGE layers ----
    for (int l = 0; l < L; ++l) {
        const size_t wo = (size_t)l * 512 * 512;
        const size_t bo = (size_t)l * 512;
        // near: window -> window
        seg_mean(h_win, off_nr, src_nr, NW, T1);
        gemm(T1, 0, wt_nl + wo, b_near + bo, o_win, NW, 512, 512, 512, 0, 0);
        gemm(h_win, 0, wt_nr + wo, nullptr, o_win, NW, 512, 512, 512, 1, 0);
        // refer: example -> window (accumulate into o_win)
        seg_mean(h_exp, off_rf, src_rf, NW, T1);
        gemm(T1, 0, wt_rl + wo, b_refer + bo, o_win, NW, 512, 512, 512, 1, 0);
        gemm(h_win, 0, wt_rr + wo, nullptr, o_win, NW, 512, 512, 512, 1, 0);
        // close: example -> example
        seg_mean(h_exp, off_cl, src_cl, NE, T1);
        gemm(T1, 0, wt_cl + wo, b_close + bo, o_cls, NE, 512, 512, 512, 0, 0);
        gemm(h_exp, 0, wt_cr + wo, nullptr, o_cls, NE, 512, 512, 512, 1, 0);
        // activations (HeteroConv mean over 2 relations for window)
        scale_lrelu_k<<<dim3((unsigned)cdiv((long long)NW * Hn, 256)), dim3(256), 0, stream>>>(
            h_win, o_win, 0.5f, (long long)NW * Hn);
        scale_lrelu_k<<<dim3((unsigned)cdiv((long long)NE * Hn, 256)), dim3(256), 0, stream>>>(
            h_exp, o_cls, 1.0f, (long long)NE * Hn);
    }

    // ---- CSRA pooling over refer edges (gather form) ----
    score_rows_k<<<dim3((unsigned)cdiv(NE, 8)), dim3(256), 0, stream>>>(h_exp, w_pool, score, NE);
    csra_att_k<<<dim3((unsigned)cdiv(NW, 256)), dim3(256), 0, stream>>>(score, off_rf, src_rf, attE, NW);
    pooled_gather_k<<<dim3((unsigned)NW), dim3(128), 0, stream>>>(h_exp, off_rf, src_rf, attE, pooled);
    // out_pre = h_win + LAM * pooled
    axpy_k<<<dim3((unsigned)cdiv((long long)NW * Hn, 256)), dim3(256), 0, stream>>>(
        T1, h_win, pooled, LAMW, (long long)NW * Hn);

    // ---- final linear -> d_out [NW x 100] fp32 ----
    gemm(T1, 0, wt_lin, b_lin, (float*)d_out, NW, 100, 512, 512, 0, 0);
    (void)out_size;
}
